// TransformerEncoder_25555055411784
// MI455X (gfx1250) — matrix operand order repaired and checked by reading, not compiled, not run on hardware
//
#include <hip/hip_runtime.h>
#include <hip/hip_bf16.h>
#include <math.h>

// ---------------- CDNA5 WMMA types / helpers ----------------
typedef __attribute__((ext_vector_type(16))) __bf16 v16bf;
typedef __attribute__((ext_vector_type(8)))  float  v8f;

#define L_SEQ 2048
#define BS    8
#define EMB   512
#define NH    8
#define DH    64
#define BH    64      // BS*NH
#define N3E   1536

__device__ __forceinline__ v8f wmma_bf16(v16bf a, v16bf b, v8f c) {
  // D = A(16x32 bf16) * B(32x16 bf16) + C(16x16 f32)
  return __builtin_amdgcn_wmma_f32_16x16x32_bf16(false, a, false, b, (short)0, c,
                                                 false, false);
}

// Generic LDS pointer low 32 bits == LDS byte offset (aperture in high bits).
__device__ __forceinline__ unsigned lds_addr32(const void* p) {
  return (unsigned)(unsigned long long)p;
}
// CDNA5 async direct global->LDS copies (ASYNCcnt tracked, no VGPR round trip)
__device__ __forceinline__ void async_ld_b64(unsigned lds, const void* g) {
  asm volatile("global_load_async_to_lds_b64 %0, %1, off"
               :: "v"(lds), "v"((unsigned long long)g) : "memory");
}
__device__ __forceinline__ void async_ld_b128(unsigned lds, const void* g) {
  asm volatile("global_load_async_to_lds_b128 %0, %1, off"
               :: "v"(lds), "v"((unsigned long long)g) : "memory");
}
__device__ __forceinline__ void wait_async0() {
  asm volatile("s_wait_asynccnt 0x0" ::: "memory");
}

// A-matrix 16x32 bf16 fragment (ISA 7.12.2): lane m=lane&15, hi=lane>>4
//   elements 0..7  -> K = hi*8 + e        (contiguous 16B)
//   elements 8..15 -> K = 16 + hi*8 + e   (contiguous 16B)
__device__ __forceinline__ v16bf load_A(const __bf16* tile, int ldk, int lane) {
  int m = lane & 15, hi = lane >> 4;
  const unsigned long long* p0 = (const unsigned long long*)(tile + m * ldk + hi * 8);
  const unsigned long long* p1 = (const unsigned long long*)(tile + m * ldk + 16 + hi * 8);
  union { v16bf v; unsigned long long q[4]; } u;
  u.q[0] = p0[0]; u.q[1] = p0[1]; u.q[2] = p1[0]; u.q[3] = p1[1];
  return u.v;
}

// B-matrix 32x16 bf16 fragment from a [N][ldk] (row = output col, K contiguous)
// lane n=lane&15 is the column; elements 0..7 -> K = hi*8 + e, 8..15 -> K = 16 + hi*8 + e
__device__ __forceinline__ v16bf load_Bnk(const __bf16* tile, int ldk, int lane) {
  int n = lane & 15, hi = lane >> 4;
  const unsigned long long* p = (const unsigned long long*)(tile + n * ldk + hi * 8);
  const unsigned long long* p1 = (const unsigned long long*)(tile + n * ldk + 16 + hi * 8);
  union { v16bf v; unsigned long long q[4]; } u;
  u.q[0] = p[0]; u.q[1] = p[1]; u.q[2] = p1[0]; u.q[3] = p1[1];
  return u.v;
}

// ---------------- 1. weight f32 -> bf16 ----------------
__global__ void convert_w_kernel(const float* __restrict__ w, __bf16* __restrict__ wb, int n) {
  int i = blockIdx.x * 256 + threadIdx.x;
  if (i < n) wb[i] = (__bf16)w[i];
}

// ---------------- 2. embedding + positional, bf16 X + f32 column sums -------
// grid (4 e-chunks, 8 batch, 16 l-chunks), block 128
__global__ void embed_kernel(const int* __restrict__ inputs, const float* __restrict__ table,
                             __bf16* __restrict__ embb, float* __restrict__ embsum) {
  int e  = blockIdx.x * 128 + threadIdx.x;  // 0..511
  int b  = blockIdx.y;
  int lc = blockIdx.z;
  int j  = e >> 1;
  float f = __expf((float)j * (-9.210340371976184f / 256.0f));  // 10000^(-j/256)
  float acc = 0.f;
  for (int l = lc * 128; l < lc * 128 + 128; ++l) {
    int tok  = inputs[b * L_SEQ + l];
    float v  = table[(size_t)tok * EMB + e];
    float ph = (float)l * f;
    float pv = (e & 1) ? __cosf(ph) : __sinf(ph);
    float val = v + pv;
    acc += val;
    embb[((size_t)l * BS + b) * EMB + e] = (__bf16)val;
  }
  atomicAdd(&embsum[b * EMB + e], acc);
}

// ---------------- 3. QKV GEMM (bf16 WMMA) ----------------
// X [16384,512] bf16 * W^T (W [1536,512] bf16) + bias, scatter into Q/K/V
// grid (24 n-tiles, 128 m-tiles), block 256 (8 waves), tile 128M x 64N
__global__ void __launch_bounds__(256) qkv_gemm_kernel(
    const __bf16* __restrict__ X, const __bf16* __restrict__ W,
    const float* __restrict__ bias, __bf16* __restrict__ Q,
    __bf16* __restrict__ K, __bf16* __restrict__ V) {
  __shared__ __bf16 Xs[128][40];
  __shared__ __bf16 Ws[64][40];
  int t = threadIdx.x, lane = t & 31, w = t >> 5;
  int hi = lane >> 4, nn = lane & 15;
  int n0 = blockIdx.x * 64;
  int m0 = blockIdx.y * 128;

  v8f c[4];
#pragma unroll
  for (int j = 0; j < 4; ++j)
#pragma unroll
    for (int i = 0; i < 8; ++i) c[j][i] = 0.f;

  for (int kb = 0; kb < 16; ++kb) {
    {  // async-stage X tile 128x32 (2 threads/row, 2x16B) + W tile 64x32 (16B)
      int row = t >> 1, seg = t & 1;
      const __bf16* gx = X + (size_t)(m0 + row) * EMB + kb * 32 + seg * 16;
      async_ld_b128(lds_addr32(&Xs[row][seg * 16]),     gx);
      async_ld_b128(lds_addr32(&Xs[row][seg * 16 + 8]), gx + 8);
      int row2 = t >> 2, quad = t & 3;
      async_ld_b128(lds_addr32(&Ws[row2][quad * 8]),
                    W + (size_t)(n0 + row2) * EMB + kb * 32 + quad * 8);
    }
    wait_async0();
    __syncthreads();
    v16bf a = load_A(&Xs[w * 16][0], 40, lane);
#pragma unroll
    for (int j = 0; j < 4; ++j) {
      v16bf bb = load_Bnk(&Ws[j * 16][0], 40, lane);
      c[j] = wmma_bf16(a, bb, c[j]);
    }
    __syncthreads();
  }
  // epilogue: n -> (head, q/k/v, d); m -> (l, b);  Q gets *0.125 (attn scale)
#pragma unroll
  for (int j = 0; j < 4; ++j) {
    int n = n0 + j * 16 + nn;
    float bv = bias[n];
    int hd = n / 192, rem = n % 192, tpe = rem >> 6, d = rem & 63;
#pragma unroll
    for (int r = 0; r < 8; ++r) {
      int m = m0 + w * 16 + hi * 8 + r;
      int l = m >> 3, b = m & 7;
      size_t off = ((size_t)(b * NH + hd) * L_SEQ + l) * DH + d;
      float val = c[j][r] + bv;
      if (tpe == 0)      Q[off] = (__bf16)(val * 0.125f);
      else if (tpe == 1) K[off] = (__bf16)val;
      else               V[off] = (__bf16)val;
    }
  }
}

// ---------------- 4. attention pass A: per-row online (max, sumexp) --------
// grid (16 q-tiles of 128, 64 bh), block 256 (8 waves, wave = 16 q rows)
__global__ void __launch_bounds__(256) attn_ml_kernel(
    const __bf16* __restrict__ Q, const __bf16* __restrict__ K,
    float* __restrict__ m_g, float* __restrict__ rl_g) {
  __shared__ __bf16 Ks[16][72];
  int t = threadIdx.x, lane = t & 31, w = t >> 5, hi = lane >> 4;
  int q0 = blockIdx.x * 128;
  int bh = blockIdx.y;

  const __bf16* Qb = Q + ((size_t)bh * L_SEQ + q0 + w * 16) * DH;
  v16bf a0 = load_A(Qb, DH, lane);        // d 0..31
  v16bf a1 = load_A(Qb + 32, DH, lane);   // d 32..63

  float mr[8], lr[8];
#pragma unroll
  for (int r = 0; r < 8; ++r) { mr[r] = -1e30f; lr[r] = 0.f; }

  int row = t >> 4, oct = t & 15;  // staging assignment: 8B per thread
  for (int kt = 0; kt < 128; ++kt) {
    async_ld_b64(lds_addr32(&Ks[row][oct * 4]),
                 K + ((size_t)bh * L_SEQ + kt * 16 + row) * DH + oct * 4);
    wait_async0();
    __syncthreads();
    v16bf b0 = load_Bnk(&Ks[0][0], 72, lane);
    v16bf b1 = load_Bnk(&Ks[0][32], 72, lane);
    v8f c;
#pragma unroll
    for (int i = 0; i < 8; ++i) c[i] = 0.f;
    c = wmma_bf16(a0, b0, c);
    c = wmma_bf16(a1, b1, c);
    __syncthreads();
    // per-lane online softmax stats over this lane's column slice
#pragma unroll
    for (int r = 0; r < 8; ++r) {
      float s   = c[r];
      float val = (s == 0.f) ? -1e30f : s;   // mask: score==0 excluded
      float mn  = fmaxf(mr[r], val);
      lr[r] = lr[r] * __expf(mr[r] - mn) + __expf(val - mn);
      mr[r] = mn;
    }
  }
  // merge (m,l) across the 16 lanes holding each row
#pragma unroll
  for (int r = 0; r < 8; ++r) {
    for (int mk = 1; mk < 16; mk <<= 1) {
      float mo = __shfl_xor(mr[r], mk);
      float lo = __shfl_xor(lr[r], mk);
      float mn = fmaxf(mr[r], mo);
      lr[r] = lr[r] * __expf(mr[r] - mn) + lo * __expf(mo - mn);
      mr[r] = mn;
    }
  }
  if ((lane & 15) == 0) {
    int qrow = q0 + w * 16 + hi * 8;
    for (int r = 0; r < 8; ++r) {
      m_g[(size_t)bh * L_SEQ + qrow + r]  = mr[r];
      rl_g[(size_t)bh * L_SEQ + qrow + r] = 1.f / lr[r];
    }
  }
}

// ---------------- 5. attention pass B: aw[k] = sum_q p(q,k) ----------------
// grid (16 k-tiles of 128, 64 bh), block 256 (8 waves, wave = 16 k cols)
__global__ void __launch_bounds__(256) attn_aw_kernel(
    const __bf16* __restrict__ Q, const __bf16* __restrict__ K,
    const float* __restrict__ m_g, const float* __restrict__ rl_g,
    float* __restrict__ aw) {
  int t = threadIdx.x, lane = t & 31, w = t >> 5, hi = lane >> 4, nn = lane & 15;
  int k0 = blockIdx.x * 128;
  int bh = blockIdx.y;

  const __bf16* Kb = K + ((size_t)bh * L_SEQ + k0 + w * 16) * DH;
  v16bf b0 = load_Bnk(Kb, DH, lane);       // fixed for whole q loop
  v16bf b1 = load_Bnk(Kb + 32, DH, lane);

  float acc = 0.f;
  for (int qt = 0; qt < 128; ++qt) {
    const __bf16* Qb = Q + ((size_t)bh * L_SEQ + qt * 16) * DH;
    v16bf a0 = load_A(Qb, DH, lane);
    v16bf a1 = load_A(Qb + 32, DH, lane);
    v8f c;
#pragma unroll
    for (int i = 0; i < 8; ++i) c[i] = 0.f;
    c = wmma_bf16(a0, b0, c);
    c = wmma_bf16(a1, b1, c);

    int mb = bh * L_SEQ + qt * 16 + hi * 8;
    float4 mq0 = *(const float4*)(m_g + mb);
    float4 mq1 = *(const float4*)(m_g + mb + 4);
    float4 rq0 = *(const float4*)(rl_g + mb);
    float4 rq1 = *(const float4*)(rl_g + mb + 4);
    float mqa[8] = {mq0.x, mq0.y, mq0.z, mq0.w, mq1.x, mq1.y, mq1.z, mq1.w};
    float rqa[8] = {rq0.x, rq0.y, rq0.z, rq0.w, rq1.x, rq1.y, rq1.z, rq1.w};
#pragma unroll
    for (int r = 0; r < 8; ++r) {
      float s = c[r];
      float p = (s == 0.f) ? 0.f : __expf(s - mqa[r]) * rqa[r];
      acc += p;
    }
  }
  acc += __shfl_xor(acc, 16);  // combine rows 0-7 half with rows 8-15 half
  if (hi == 0) aw[(size_t)bh * L_SEQ + k0 + w * 16 + nn] = acc;
}

// ---------------- 6. attn_out[bh][d] = sum_k aw[bh][k] * V[bh][k][d] -------
__global__ void av_kernel(const float* __restrict__ aw, const __bf16* __restrict__ V,
                          float* __restrict__ attn) {
  __shared__ float sm[256];
  int bh = blockIdx.x, t = threadIdx.x;
  int d = t & 63, part = t >> 6;
  float acc = 0.f;
  for (int k = part; k < L_SEQ; k += 4)
    acc += aw[(size_t)bh * L_SEQ + k] * (float)V[((size_t)bh * L_SEQ + k) * DH + d];
  sm[t] = acc;
  __syncthreads();
  if (t < 64) attn[bh * DH + t] = sm[t] + sm[t + 64] + sm[t + 128] + sm[t + 192];
}

// ---------------- 7. GroupNorm + residual + LayerNorm + FFN ----------------
// one block per batch, 512 threads (one per channel)
__global__ void __launch_bounds__(512) final_kernel(
    const float* __restrict__ attn, const float* __restrict__ embsum,
    const int* __restrict__ lens, const float* __restrict__ gn_w,
    const float* __restrict__ gn_b, const float* __restrict__ ln_w,
    const float* __restrict__ ln_b, const float* __restrict__ w1,
    const float* __restrict__ b1, const float* __restrict__ w2,
    const float* __restrict__ b2, float* __restrict__ out) {
  int b = blockIdx.x, t = threadIdx.x;
  __shared__ float ws1[16], ws2[16], xn[512], h1[1024], sm[2];

  float a = attn[b * EMB + t];  // layout (b*8+h)*64+d == b*512 + t
  float s1 = a, s2 = a * a;
  for (int m = 1; m < 32; m <<= 1) { s1 += __shfl_xor(s1, m); s2 += __shfl_xor(s2, m); }
  int wave = t >> 5;
  if ((t & 31) == 0) { ws1[wave] = s1; ws2[wave] = s2; }
  __syncthreads();
  int h = t >> 6;
  float gs = ws1[2 * h] + ws1[2 * h + 1];
  float gq = ws2[2 * h] + ws2[2 * h + 1];
  float mu = gs / 64.f, var = gq / 64.f - mu * mu;
  float gn = (a - mu) * rsqrtf(var + 1e-5f) * gn_w[h] + gn_b[h];
  float x = gn + embsum[b * EMB + t] / (float)lens[b];
  __syncthreads();

  s1 = x; s2 = x * x;
  for (int m = 1; m < 32; m <<= 1) { s1 += __shfl_xor(s1, m); s2 += __shfl_xor(s2, m); }
  if ((t & 31) == 0) { ws1[wave] = s1; ws2[wave] = s2; }
  __syncthreads();
  if (t == 0) {
    float a1 = 0.f, a2 = 0.f;
    for (int i = 0; i < 16; ++i) { a1 += ws1[i]; a2 += ws2[i]; }
    sm[0] = a1 / 512.f;
    sm[1] = a2 / 512.f;
  }
  __syncthreads();
  float mu2 = sm[0], var2 = sm[1] - mu2 * mu2;
  xn[t] = (x - mu2) * rsqrtf(var2 + 1e-5f) * ln_w[t] + ln_b[t];
  __syncthreads();

  for (int jj = t; jj < 1024; jj += 512) {
    float acc = b1[jj];
    const float* wr = w1 + (size_t)jj * 512;
    for (int e = 0; e < 512; ++e) acc += xn[e] * wr[e];
    h1[jj] = tanhf(acc);
  }
  __syncthreads();
  float acc = b2[t];
  const float* wr = w2 + (size_t)t * 1024;
  for (int j = 0; j < 1024; ++j) acc += h1[j] * wr[j];
  out[b * EMB + t] = x + acc;
}

// ---------------- host launcher ----------------
extern "C" void kernel_launch(void* const* d_in, const int* in_sizes, int n_in,
                              void* d_out, int out_size, void* d_ws, size_t ws_size,
                              hipStream_t stream) {
  (void)in_sizes; (void)n_in; (void)out_size; (void)ws_size;
  const int*   inputs  = (const int*)d_in[0];
  const int*   lens    = (const int*)d_in[1];
  const float* emb_tab = (const float*)d_in[2];
  const float* qkv_w   = (const float*)d_in[3];
  const float* qkv_b   = (const float*)d_in[4];
  const float* gn_w    = (const float*)d_in[5];
  const float* gn_b    = (const float*)d_in[6];
  const float* ln_w    = (const float*)d_in[7];
  const float* ln_b    = (const float*)d_in[8];
  const float* w1      = (const float*)d_in[9];
  const float* b1      = (const float*)d_in[10];
  const float* w2      = (const float*)d_in[11];
  const float* b2      = (const float*)d_in[12];
  float* out = (float*)d_out;

  char* ws = (char*)d_ws;
  size_t off = 0;
  auto alloc = [&](size_t bytes) -> void* {
    void* p = ws + off;
    off = (off + bytes + 255) & ~(size_t)255;
    return p;
  };
  __bf16* embb  = (__bf16*)alloc((size_t)16384 * 512 * 2);
  __bf16* wb    = (__bf16*)alloc((size_t)1536 * 512 * 2);
  __bf16* Q     = (__bf16*)alloc((size_t)BH * L_SEQ * DH * 2);
  __bf16* Km    = (__bf16*)alloc((size_t)BH * L_SEQ * DH * 2);
  __bf16* V     = (__bf16*)alloc((size_t)BH * L_SEQ * DH * 2);
  float*  m_g   = (float*)alloc((size_t)BH * L_SEQ * 4);
  float*  rl_g  = (float*)alloc((size_t)BH * L_SEQ * 4);
  float*  aw    = (float*)alloc((size_t)BH * L_SEQ * 4);
  float*  esum  = (float*)alloc((size_t)BS * EMB * 4);
  float*  attn  = (float*)alloc((size_t)BH * DH * 4);

  hipMemsetAsync(esum, 0, (size_t)BS * EMB * 4, stream);

  convert_w_kernel<<<3072, 256, 0, stream>>>(qkv_w, wb, 1536 * 512);
  embed_kernel<<<dim3(4, 8, 16), 128, 0, stream>>>(inputs, emb_tab, embb, esum);
  qkv_gemm_kernel<<<dim3(24, 128), 256, 0, stream>>>(embb, wb, qkv_b, Q, Km, V);
  attn_ml_kernel<<<dim3(16, 64), 256, 0, stream>>>(Q, Km, m_g, rl_g);
  attn_aw_kernel<<<dim3(16, 64), 256, 0, stream>>>(Q, Km, m_g, rl_g, aw);
  av_kernel<<<64, 256, 0, stream>>>(aw, V, attn);
  final_kernel<<<8, 512, 0, stream>>>(attn, esum, lens, gn_w, gn_b, ln_w, ln_b,
                                      w1, b1, w2, b2, out);
}
